// UltraSparseMoE_78185584656907
// MI455X (gfx1250) — compile-verified, matrix operand order checked by reading, not run on hardware
//
#include <hip/hip_runtime.h>
#include <hip/hip_bf16.h>
#include <stdint.h>

// ---------------- problem constants ----------------
#define TB 2
#define TS 1024
#define TD 1024          // model dim D
#define TH 4096          // SwiGLU hidden H
#define TE 8             // routed experts
#define TT (TB * TS)     // tokens = 2048

// tile config: 256 threads = 8 wave32; block tile 64x64, wave tile 16x32, K-step 32
#define BM 64
#define BN 64
#define BK 32
#define LDW 17           // padded LDS stride in u32 (17 banks -> conflict-spread)
#define ALDW 20          // stride for async-staged tile: 16B-aligned rows,
                         // 20*r mod 64 distinct for r in 0..15 -> conflict-free frags

#define AS1 __attribute__((address_space(1)))
#define AS3 __attribute__((address_space(3)))

#if defined(__has_builtin)
#if __has_builtin(__builtin_amdgcn_global_load_async_to_lds_b128) && \
    __has_builtin(__builtin_amdgcn_s_wait_asynccnt)
#define USE_ASYNC_LDS 1
#endif
#endif

typedef __bf16 bf16;
typedef __attribute__((ext_vector_type(16))) __bf16    v16bf;
typedef __attribute__((ext_vector_type(8)))  float     v8f;
typedef __attribute__((ext_vector_type(8)))  uint32_t  v8u;
typedef __attribute__((ext_vector_type(4)))  int       v4i;

static __device__ __forceinline__ bf16 f2bf(float f) {
    uint32_t u = __builtin_bit_cast(uint32_t, f);
    u += 0x7FFFu + ((u >> 16) & 1u);              // round-to-nearest-even
    uint16_t h = (uint16_t)(u >> 16);
    return __builtin_bit_cast(bf16, h);
}

// pack two fp32 -> {bf16 lo, bf16 hi} in one dword (RNE)
static __device__ __forceinline__ uint32_t pk2bf(float lo, float hi) {
    uint32_t a = __builtin_bit_cast(uint32_t, lo);
    a += 0x7FFFu + ((a >> 16) & 1u);
    uint32_t b = __builtin_bit_cast(uint32_t, hi);
    b += 0x7FFFu + ((b >> 16) & 1u);
    return (a >> 16) | (b & 0xFFFF0000u);
}

// gather a 16x32 bf16 fragment from LDS tile stored as packed u32 [row][k/2]
// lane l: row' = l&15, K base = (l>>4)*8 ; v16bf elem pair (2i,2i+1) = u32 i
static __device__ __forceinline__ v16bf frag_ld(const uint32_t* tile, int row,
                                                int kw, int stride) {
    const uint32_t* p = tile + row * stride + kw;
    v8u u;
#pragma unroll
    for (int i = 0; i < 8; ++i) u[i] = p[(i & 3) + ((i >> 2) * 8)];
    return __builtin_bit_cast(v16bf, u);
}

// ---------------- router: logits, softmax, top-2, expert scatter ----------------
__global__ __launch_bounds__(256)
void moe_router_kernel(const float* __restrict__ x, const float* __restrict__ rw,
                       float* __restrict__ logits_out,
                       int* __restrict__ cnt, int* __restrict__ toklist,
                       float* __restrict__ tokw) {
    __shared__ float rwl[TE * TD];                // 32 KB
    int tid = threadIdx.x;
    for (int i = tid; i < TE * TD; i += 256) rwl[i] = rw[i];
    __syncthreads();

    int t = blockIdx.x * 256 + tid;               // one thread per token
    float acc[TE];
#pragma unroll
    for (int e = 0; e < TE; ++e) acc[e] = 0.f;

    const float* xr = x + (size_t)t * TD;
    for (int d = 0; d < TD; ++d) {
        float xv = xr[d];
#pragma unroll
        for (int e = 0; e < TE; ++e) acc[e] += xv * rwl[e * TD + d];
    }
#pragma unroll
    for (int e = 0; e < TE; ++e) logits_out[t * TE + e] = acc[e];

    // softmax
    float m = acc[0];
#pragma unroll
    for (int e = 1; e < TE; ++e) m = fmaxf(m, acc[e]);
    float p[TE], s = 0.f;
#pragma unroll
    for (int e = 0; e < TE; ++e) { p[e] = __expf(acc[e] - m); s += p[e]; }
    float inv = 1.f / s;
#pragma unroll
    for (int e = 0; e < TE; ++e) p[e] *= inv;

    // top-2 (first-index tie-break, like lax.top_k)
    int i1 = 0;
#pragma unroll
    for (int e = 1; e < TE; ++e) if (p[e] > p[i1]) i1 = e;
    int i2 = (i1 == 0) ? 1 : 0;
#pragma unroll
    for (int e = 0; e < TE; ++e) if (e != i1 && p[e] > p[i2]) i2 = e;
    float ws = p[i1] + p[i2];
    float w1 = p[i1] / ws, w2 = p[i2] / ws;

    int s1 = atomicAdd(&cnt[i1], 1);
    toklist[i1 * TT + s1] = t; tokw[i1 * TT + s1] = w1;
    int s2 = atomicAdd(&cnt[i2], 1);
    toklist[i2 * TT + s2] = t; tokw[i2 * TT + s2] = w2;
}

__global__ void moe_zero_cnt_kernel(int* cnt) {
    if (threadIdx.x < TE) cnt[threadIdx.x] = 0;
}

// ---------------- pass 1 GEMM: hbuf = silu(X@W1) * (X@W2), bf16 out ----------------
// X fp32 [TT, TD] (rows gathered via toklist when routed); W1,W2 fp32 [TD, TH]
__global__ __launch_bounds__(256)
void moe_swiglu_gemm(const float* __restrict__ X,
                     const float* __restrict__ W1, const float* __restrict__ W2,
                     bf16* __restrict__ hbuf,
                     const int* __restrict__ toklist, const int* __restrict__ cntp) {
    int cnt = cntp ? *cntp : TT;
    int m0 = blockIdx.y * BM;
    if (m0 >= cnt) return;
    int n0 = blockIdx.x * BN;

    __shared__ uint32_t As[BM * LDW];     // [m][k/2] packed bf16 pairs
    __shared__ uint32_t B1s[BN * LDW];    // [n][k/2] (transposed)
    __shared__ uint32_t B2s[BN * LDW];

    int tid  = threadIdx.x;
    int lane = tid & 31;
    int wid  = tid >> 5;
    int wm   = wid & 3;                   // 0..3 -> M subtile
    int wn   = wid >> 2;                  // 0..1 -> N slab of 32

    // staging coordinates
    int ar  = tid >> 2;                   // A row 0..63
    int acw = (tid & 3) * 4;              // A u32 col 0,4,8,12
    int bkp = tid >> 4;                   // B k-pair 0..15 (k = 2*bkp)
    int bc  = (tid & 15) * 4;             // B n col 0..60

    // fragment coordinates
    int lrow = lane & 15;
    int kw   = (lane >> 4) * 4;           // u32 K base (halves base = 8)
    int arow = wm * 16 + lrow;

    v8f acc1[2] = {}, acc2[2] = {};

    for (int k0 = 0; k0 < TD; k0 += BK) {
        __syncthreads();
        // ---- stage A: 64x32 fp32 -> packed bf16 pairs (4 dword stores) ----
        {
            int gm = m0 + ar;
            float v[8];
            if (gm < cnt) {
                int row = toklist ? toklist[gm] : gm;
                const float* p = X + (size_t)row * TD + k0 + acw * 2;
#pragma unroll
                for (int j = 0; j < 8; ++j) v[j] = p[j];
            } else {
#pragma unroll
                for (int j = 0; j < 8; ++j) v[j] = 0.f;
            }
#pragma unroll
            for (int j = 0; j < 4; ++j)
                As[ar * LDW + acw + j] = pk2bf(v[2 * j], v[2 * j + 1]);
        }
        // ---- stage B1/B2: rows (k,k+1) x 4 cols -> packed, transposed ----
        {
            size_t r0 = (size_t)(k0 + 2 * bkp) * TH + n0 + bc;
            const float* a1 = W1 + r0;  const float* b1 = a1 + TH;
            const float* a2 = W2 + r0;  const float* b2 = a2 + TH;
#pragma unroll
            for (int j = 0; j < 4; ++j) {
                B1s[(bc + j) * LDW + bkp] = pk2bf(a1[j], b1[j]);
                B2s[(bc + j) * LDW + bkp] = pk2bf(a2[j], b2[j]);
            }
            if (k0 + BK < TD) {           // prefetch next K-step weight rows
                __builtin_prefetch(a1 + (size_t)BK * TH, 0, 1);
                __builtin_prefetch(a2 + (size_t)BK * TH, 0, 1);
            }
        }
        __syncthreads();

        // ---- fragments first, then 4 back-to-back WMMAs ----
        v16bf a   = frag_ld(As,  arow, kw, LDW);
        v16bf b10 = frag_ld(B1s, wn * 32 + lrow,      kw, LDW);
        v16bf b11 = frag_ld(B1s, wn * 32 + 16 + lrow, kw, LDW);
        v16bf b20 = frag_ld(B2s, wn * 32 + lrow,      kw, LDW);
        v16bf b21 = frag_ld(B2s, wn * 32 + 16 + lrow, kw, LDW);

        acc1[0] = __builtin_amdgcn_wmma_f32_16x16x32_bf16(
            false, a, false, b10, (short)0, acc1[0], false, false);
        acc1[1] = __builtin_amdgcn_wmma_f32_16x16x32_bf16(
            false, a, false, b11, (short)0, acc1[1], false, false);
        acc2[0] = __builtin_amdgcn_wmma_f32_16x16x32_bf16(
            false, a, false, b20, (short)0, acc2[0], false, false);
        acc2[1] = __builtin_amdgcn_wmma_f32_16x16x32_bf16(
            false, a, false, b21, (short)0, acc2[1], false, false);
    }

    // ---- epilogue: h = silu(g) * u, store bf16 to compacted hbuf ----
    int lcol  = lane & 15;
    int rbase = (lane >> 4) * 8;
#pragma unroll
    for (int s = 0; s < 2; ++s) {
#pragma unroll
        for (int r = 0; r < 8; ++r) {
            float g = acc1[s][r];
            float u = acc2[s][r];
            float h = (g / (1.f + __expf(-g))) * u;
            int mrow = m0 + wm * 16 + rbase + r;
            if (mrow < cnt)
                hbuf[(size_t)mrow * TH + n0 + wn * 32 + s * 16 + lcol] = f2bf(h);
        }
    }
}

// ---------------- pass 2 GEMM: out[tok] (+)= w * (hbuf @ W3) ----------------
// hbuf bf16 [cnt, TH] compacted; W3 fp32 [TH, TD]
__global__ __launch_bounds__(256)
void moe_out_gemm(const bf16* __restrict__ hbuf, const float* __restrict__ W3,
                  float* __restrict__ out,
                  const int* __restrict__ toklist, const int* __restrict__ cntp,
                  const float* __restrict__ tokw, int accumulate) {
    int cnt = cntp ? *cntp : TT;
    int m0 = blockIdx.y * BM;
    if (m0 >= cnt) return;
    int n0 = blockIdx.x * BN;

    __shared__ uint32_t As[BM * ALDW];    // async-staged: 16B-aligned rows
    __shared__ uint32_t Bs[BN * LDW];

    int tid  = threadIdx.x;
    int lane = tid & 31;
    int wid  = tid >> 5;
    int wm   = wid & 3;
    int wn   = wid >> 2;

    int ar  = tid >> 2;
    int acw = (tid & 3) * 4;
    int bkp = tid >> 4;
    int bc  = (tid & 15) * 4;

    int lrow = lane & 15;
    int kw   = (lane >> 4) * 4;
    int arow = wm * 16 + lrow;

    v8f acc[2] = {};

    for (int k0 = 0; k0 < TH; k0 += BK) {
        __syncthreads();
        {   // stage A: hbuf already bf16 -> async DMA 16B/thread into LDS
            int gm = m0 + ar;
            uint32_t* dst = &As[ar * ALDW + acw];
            if (gm < cnt) {
                const bf16* src = hbuf + (size_t)gm * TH + k0 + acw * 2;
#ifdef USE_ASYNC_LDS
                __builtin_amdgcn_global_load_async_to_lds_b128(
                    (AS1 v4i*)src, (AS3 v4i*)dst, 0, 0);
#else
                const uint32_t* p = (const uint32_t*)src;
#pragma unroll
                for (int j = 0; j < 4; ++j) dst[j] = p[j];
#endif
            } else {
#pragma unroll
                for (int j = 0; j < 4; ++j) dst[j] = 0u;
            }
        }
        {   // stage B: W3 rows (k,k+1) -> packed bf16, transposed
            size_t r0 = (size_t)(k0 + 2 * bkp) * TD + n0 + bc;
            const float* pa = W3 + r0;
            const float* pb = pa + TD;
#pragma unroll
            for (int j = 0; j < 4; ++j)
                Bs[(bc + j) * LDW + bkp] = pk2bf(pa[j], pb[j]);
            if (k0 + BK < TH)
                __builtin_prefetch(pa + (size_t)BK * TD, 0, 1);
        }
#ifdef USE_ASYNC_LDS
        __builtin_amdgcn_s_wait_asynccnt(0);   // async A-tile landed in LDS
#endif
        __syncthreads();

        v16bf a  = frag_ld(As, arow, kw, ALDW);
        v16bf b0 = frag_ld(Bs, wn * 32 + lrow,      kw, LDW);
        v16bf b1 = frag_ld(Bs, wn * 32 + 16 + lrow, kw, LDW);

        acc[0] = __builtin_amdgcn_wmma_f32_16x16x32_bf16(
            false, a, false, b0, (short)0, acc[0], false, false);
        acc[1] = __builtin_amdgcn_wmma_f32_16x16x32_bf16(
            false, a, false, b1, (short)0, acc[1], false, false);
    }

    int lcol  = lane & 15;
    int rbase = (lane >> 4) * 8;
#pragma unroll
    for (int s = 0; s < 2; ++s) {
#pragma unroll
        for (int r = 0; r < 8; ++r) {
            int slot = m0 + wm * 16 + rbase + r;
            if (slot < cnt) {
                int t   = toklist ? toklist[slot] : slot;
                float w = tokw ? tokw[slot] : 1.f;
                size_t o = (size_t)t * TD + n0 + wn * 32 + s * 16 + lcol;
                float v = w * acc[s][r];
                if (accumulate) out[o] += v;   // stream-ordered passes: no races
                else            out[o]  = v;
            }
        }
    }
}

// ---------------- launch ----------------
extern "C" void kernel_launch(void* const* d_in, const int* in_sizes, int n_in,
                              void* d_out, int out_size, void* d_ws, size_t ws_size,
                              hipStream_t stream) {
    (void)in_sizes; (void)n_in; (void)out_size; (void)ws_size;
    const float* x   = (const float*)d_in[0];
    const float* sw1 = (const float*)d_in[1];   // [1, D, H]
    const float* sw2 = (const float*)d_in[2];   // [1, D, H]
    const float* sw3 = (const float*)d_in[3];   // [1, H, D]
    const float* w1  = (const float*)d_in[4];   // [E, D, H]
    const float* w2  = (const float*)d_in[5];   // [E, D, H]
    const float* w3  = (const float*)d_in[6];   // [E, H, D]
    const float* rw  = (const float*)d_in[7];   // [E, D]

    float* out    = (float*)d_out;              // [T, D]
    float* logits = out + (size_t)TT * TD;      // [T, E]

    // workspace layout (hbuf 16B-aligned for async B128)
    int*   cnt     = (int*)d_ws;                        // E (padded to 32)
    int*   toklist = cnt + 32;                          // E*T
    float* tokw    = (float*)(toklist + TE * TT);       // E*T
    bf16*  hbuf    = (bf16*)(tokw + TE * TT);           // T*H bf16 (~16 MB)

    moe_zero_cnt_kernel<<<1, 32, 0, stream>>>(cnt);
    moe_router_kernel<<<TT / 256, 256, 0, stream>>>(x, rw, logits, cnt, toklist, tokw);

    dim3 blk(256);
    dim3 grid_h(TH / BN, TT / BM);   // 64 x 32
    dim3 grid_o(TD / BN, TT / BM);   // 16 x 32

    // shared expert: writes out (=)
    moe_swiglu_gemm<<<grid_h, blk, 0, stream>>>(x, sw1, sw2, hbuf, nullptr, nullptr);
    moe_out_gemm<<<grid_o, blk, 0, stream>>>(hbuf, sw3, out, nullptr, nullptr, nullptr, 0);

    // routed experts: gathered token lists, accumulate (+=)
    for (int e = 0; e < TE; ++e) {
        const float* e1 = w1 + (size_t)e * TD * TH;
        const float* e2 = w2 + (size_t)e * TD * TH;
        const float* e3 = w3 + (size_t)e * TH * TD;
        moe_swiglu_gemm<<<grid_h, blk, 0, stream>>>(
            x, e1, e2, hbuf, toklist + e * TT, cnt + e);
        moe_out_gemm<<<grid_o, blk, 0, stream>>>(
            hbuf, e3, out, toklist + e * TT, cnt + e, tokw + e * TT, 1);
    }
}
